// NTXentLoss_14482629722359
// MI455X (gfx1250) — matrix operand order repaired and checked by reading, not compiled, not run on hardware
//
#include <hip/hip_runtime.h>
#include <hip/hip_bf16.h>
#include <math.h>

// NT-Xent loss, fused WMMA implementation for gfx1250 (MI455X).
//
// loss = (1/N) * sum_i [ log( sum_{j != i} exp(sim_ij / T) ) - sim[i, p(i)] / T ]
// with T = 0.5, p(i) = (i + B) mod N, sim = normalized-row Gram matrix.

typedef __attribute__((ext_vector_type(16))) _Float16 v16h;
typedef __attribute__((ext_vector_type(8)))  _Float16 v8h;
typedef __attribute__((ext_vector_type(4)))  _Float16 v4h;
typedef __attribute__((ext_vector_type(8)))  float    v8f;

#define DIMD 128
#define TEMP_INV 2.0f                       // 1 / 0.5
#define EXP2_SCALE 2.8853900817779268f      // 2 * log2(e): exp(2x) == exp2(x * this)

// Async global->LDS staging of B tiles (gfx1250 ASYNCcnt path). Set to 0 to
// fall back to plain load + ds_store if the assembler rejects the mnemonic.
#ifndef NTX_USE_ASYNC_LDS
#define NTX_USE_ASYNC_LDS 1
#endif

#define LDS_STRIDE 144                  // halves per staged row (128 + 16 pad -> 288B, 32B aligned)
#define LDS_TILE   (16 * LDS_STRIDE)    // halves per 16-row tile (4608 B)
#define CHUNK_T    4                    // column tiles staged per async round
#define CHUNK_H    (CHUNK_T * LDS_TILE) // halves per chunk buffer (18432 B)

// ---------------------------------------------------------------------------
// Kernel 1: row-normalize reps = cat([zjs, zis]); emit f16 (WMMA) + f32 copies.
// One wave32 per row of 128 floats (4 per lane).
// ---------------------------------------------------------------------------
__global__ __launch_bounds__(256) void ntxent_normalize(
    const float* __restrict__ zis, const float* __restrict__ zjs,
    _Float16* __restrict__ zh, float* __restrict__ zf, int Bsz)
{
    const int row  = blockIdx.x * 8 + (threadIdx.x >> 5);
    const int lane = threadIdx.x & 31;

    const float* src = (row < Bsz) ? (zjs + (size_t)row * DIMD)
                                   : (zis + (size_t)(row - Bsz) * DIMD);
    float4 v = ((const float4*)src)[lane];

    float ss = v.x * v.x + v.y * v.y + v.z * v.z + v.w * v.w;
    ss += __shfl_xor(ss, 1, 32);
    ss += __shfl_xor(ss, 2, 32);
    ss += __shfl_xor(ss, 4, 32);
    ss += __shfl_xor(ss, 8, 32);
    ss += __shfl_xor(ss, 16, 32);

    const float inv = 1.0f / fmaxf(sqrtf(ss), 1e-8f);
    v.x *= inv; v.y *= inv; v.z *= inv; v.w *= inv;

    ((float4*)(zf + (size_t)row * DIMD))[lane] = v;
    v4h h = { (_Float16)v.x, (_Float16)v.y, (_Float16)v.z, (_Float16)v.w };
    ((v4h*)(zh + (size_t)row * DIMD))[lane] = h;
}

// ---------------------------------------------------------------------------
// Fragment loaders used here (wave32): contiguous k = 16*half + s, same for A and B.
//
// A (16x32 f16, M x K): lanes 0-15 hold M=lane, halves K0..15;
//                       lanes 16-31 hold M=lane-16, halves K16..31.
// B (32x16 f16, K x N): lanes 0-15 hold N=lane,  halves K0..15;
//                       lanes 16-31 hold N=lane-16, halves K16..31.
// ---------------------------------------------------------------------------
__device__ __forceinline__ v16h load_a_frag(const _Float16* __restrict__ p)
{
    // p points at row*128 + kk*32 + half*16 ; two 16B loads at +0, +8 halves.
    v8h lo = *(const v8h*)(p);
    v8h hi = *(const v8h*)(p + 8);
    v16h r;
#pragma unroll
    for (int t = 0; t < 8; ++t) { r[t] = lo[t]; r[t + 8] = hi[t]; }
    return r;
}

__device__ __forceinline__ void wait_async0()
{
#if NTX_USE_ASYNC_LDS
#if __has_builtin(__builtin_amdgcn_s_wait_asynccnt)
    __builtin_amdgcn_s_wait_asynccnt(0);
#else
    asm volatile("s_wait_asynccnt 0x0" ::: "memory");
#endif
#endif
}

// ---------------------------------------------------------------------------
// Kernel 2: fused Gram-matrix x exp row-sum.
// Block = 8 waves = 128 rows; each wave owns a 16-row A tile in registers.
// Chunks of 4 column tiles (64 rows x 128 f16 = 16 KB) are staged per block
// into double-buffered LDS via global_load_async_to_lds_b128 (ASYNCcnt) --
// one s_wait_asynccnt + one barrier per 16 WMMAs. Diagonal (j == i) handled
// by exact post-loop subtraction (no masking in the hot loop).
// rowsum[i] = sum_{j != i} exp(sim_ij * 2)
// ---------------------------------------------------------------------------
__global__ __launch_bounds__(256) void ntxent_rowsum(
    const _Float16* __restrict__ zh, float* __restrict__ rowsum, int N)
{
    extern __shared__ _Float16 smem[];   // 2 * CHUNK_H halves (36864 B)

    const int tid     = threadIdx.x;
    const int lane    = tid & 31;
    const int wave    = tid >> 5;
    const int rowTile = blockIdx.x * 8 + wave;
    const int rowBase = rowTile * 16;
    const int hf      = lane >> 4;   // lane half (0/1)
    const int lp      = lane & 15;   // lane-in-half == output column / A row

    // Stage one 64-row chunk (column chunk cc) into LDS buffer buf:
    // 1024 x 16B segments, 4 async B128 ops per thread.
    auto stage = [&](int cc, int buf) {
#pragma unroll
        for (int k = 0; k < 4; ++k) {
            const int s   = tid + k * 256;   // 0..1023
            const int row = s >> 4;          // 0..63
            const int seg = s & 15;          // 16B segment within 256B row
            const unsigned gOff = (unsigned)((cc * 64 + row) * DIMD + seg * 8) * 2u;
            const unsigned lOff = (unsigned)(buf * CHUNK_H + row * LDS_STRIDE + seg * 8) * 2u;
#if NTX_USE_ASYNC_LDS
            asm volatile("global_load_async_to_lds_b128 %0, %1, %2"
                         :: "v"(lOff), "v"(gOff), "s"(zh) : "memory");
#else
            float4 t = *(const float4*)((const char*)zh + gOff);
            *(float4*)((char*)smem + lOff) = t;
#endif
        }
    };

    // A fragments: 16 rows x 128 K, resident in 32 VGPRs.
    v16h a[4];
    {
        const _Float16* arow = zh + (size_t)(rowBase + lp) * DIMD + hf * 16;
#pragma unroll
        for (int kk = 0; kk < 4; ++kk) a[kk] = load_a_frag(arow + kk * 32);
    }

    float racc[8];
#pragma unroll
    for (int r = 0; r < 8; ++r) racc[r] = 0.0f;

    const int nChunks = N >> 6;              // 64 columns per chunk
    stage(0, 0);

    for (int cc = 0; cc < nChunks; ++cc) {
        wait_async0();
        __syncthreads();                     // buffer cc&1 staged; prev reads done
        if (cc + 1 < nChunks) stage(cc + 1, (cc + 1) & 1);

        const _Float16* base = smem + (cc & 1) * CHUNK_H + lp * LDS_STRIDE + hf * 16;
#pragma unroll
        for (int t = 0; t < CHUNK_T; ++t) {
            const _Float16* bp = base + t * LDS_TILE;
            v8f acc = {};
#pragma unroll
            for (int kk = 0; kk < 4; ++kk) {
                v16h b = *(const v16h*)(bp + kk * 32);   // 32B ds read
                acc = __builtin_amdgcn_wmma_f32_16x16x32_f16(
                    false, a[kk], false, b, (short)0, acc, false, false);
            }
#pragma unroll
            for (int r = 0; r < 8; ++r)
                racc[r] += exp2f(acc[r] * EXP2_SCALE);   // single mul + v_exp_f32
        }
    }

    // Exact diagonal correction: recompute the tile ct == rowTile from global
    // (bit-identical inputs -> bit-identical WMMA result) and subtract j == i.
    {
        const _Float16* brow = zh + (size_t)(rowBase + lp) * DIMD + hf * 16;
        v8f acc = {};
#pragma unroll
        for (int kk = 0; kk < 4; ++kk) {
            v16h b = *(const v16h*)(brow + kk * 32);
            acc = __builtin_amdgcn_wmma_f32_16x16x32_f16(
                false, a[kk], false, b, (short)0, acc, false, false);
        }
#pragma unroll
        for (int r = 0; r < 8; ++r)
            if (lp == (r + (hf << 3)))
                racc[r] -= exp2f(acc[r] * EXP2_SCALE);
    }

    // Reduce over the 16 lanes of each half (columns), keep halves separate.
#pragma unroll
    for (int r = 0; r < 8; ++r) {
        float s = racc[r];
        s += __shfl_xor(s, 1, 32);
        s += __shfl_xor(s, 2, 32);
        s += __shfl_xor(s, 4, 32);
        s += __shfl_xor(s, 8, 32);
        racc[r] = s;
    }
    if (lp == 0) {
#pragma unroll
        for (int r = 0; r < 8; ++r)
            rowsum[rowBase + (hf << 3) + r] = racc[r];   // exclusive owner
    }
}

// ---------------------------------------------------------------------------
// Kernel 3: per-row term = log(rowsum_i) - 2 * dot(z_i, z_{(i+B)%N}) in f32.
// One wave32 per row.
// ---------------------------------------------------------------------------
__global__ __launch_bounds__(256) void ntxent_posterm(
    const float* __restrict__ zf, const float* __restrict__ rowsum,
    float* __restrict__ term, int N, int Bsz)
{
    const int row  = blockIdx.x * 8 + (threadIdx.x >> 5);
    const int lane = threadIdx.x & 31;
    int p = row + Bsz; if (p >= N) p -= N;

    float4 x = ((const float4*)(zf + (size_t)row * DIMD))[lane];
    float4 y = ((const float4*)(zf + (size_t)p   * DIMD))[lane];
    float d = x.x * y.x + x.y * y.y + x.z * y.z + x.w * y.w;
    d += __shfl_xor(d, 1, 32);
    d += __shfl_xor(d, 2, 32);
    d += __shfl_xor(d, 4, 32);
    d += __shfl_xor(d, 8, 32);
    d += __shfl_xor(d, 16, 32);

    if (lane == 0)
        term[row] = logf(rowsum[row]) - d * TEMP_INV;
}

// ---------------------------------------------------------------------------
// Kernel 4: deterministic single-block reduction -> loss.
// ---------------------------------------------------------------------------
__global__ __launch_bounds__(256) void ntxent_reduce(
    const float* __restrict__ term, float* __restrict__ out, int N)
{
    __shared__ float s[256];
    float acc = 0.0f;
    for (int i = threadIdx.x; i < N; i += 256) acc += term[i];
    s[threadIdx.x] = acc;
    __syncthreads();
    for (int off = 128; off > 0; off >>= 1) {
        if ((int)threadIdx.x < off) s[threadIdx.x] += s[threadIdx.x + off];
        __syncthreads();
    }
    if (threadIdx.x == 0) out[0] = s[0] / (float)N;
}

// ---------------------------------------------------------------------------
extern "C" void kernel_launch(void* const* d_in, const int* in_sizes, int n_in,
                              void* d_out, int out_size, void* d_ws, size_t ws_size,
                              hipStream_t stream)
{
    const float* zis = (const float*)d_in[0];
    const float* zjs = (const float*)d_in[1];
    float* out = (float*)d_out;

    const int Bsz = in_sizes[0] / DIMD;   // 8192
    const int N   = 2 * Bsz;              // 16384

    char* ws = (char*)d_ws;
    _Float16* zh     = (_Float16*)ws;                                  // N*128*2 = 4 MB
    float*    zf     = (float*)(ws + (size_t)N * DIMD * 2);            // N*128*4 = 8 MB
    float*    rowsum = (float*)(ws + (size_t)N * DIMD * 6);            // N*4
    float*    term   = rowsum + N;                                     // N*4

    const int ldsBytes = 2 * CHUNK_H * (int)sizeof(_Float16);          // 36864 B

    ntxent_normalize<<<N / 8,   256, 0,        stream>>>(zis, zjs, zh, zf, Bsz);
    ntxent_rowsum  <<<N / 128,  256, ldsBytes, stream>>>(zh, rowsum, N);
    ntxent_posterm <<<N / 8,    256, 0,        stream>>>(zf, rowsum, term, N, Bsz);
    ntxent_reduce  <<<1,        256, 0,        stream>>>(term, out, N);
}
